// LSTMCell_30580167147733
// MI455X (gfx1250) — compile-verified, matrix operand order checked by reading, not run on hardware
//
#include <hip/hip_runtime.h>
#include <hip/hip_bf16.h>
#include <stdint.h>

#define B_DIM   8192
#define IN_DIM  1024
#define H_DIM   1024
#define K_TOT   2048          // IN + H (concatenated K)
#define TILE_K  32
#define KSTEPS  64            // 2048 / 32
#define LDS_STRIDE 40         // 32 bf16 + 8 pad -> 80 B rows (16B aligned)

typedef __bf16    v16bf  __attribute__((ext_vector_type(16)));
typedef float     v8f    __attribute__((ext_vector_type(8)));
typedef uint32_t  u32x4  __attribute__((ext_vector_type(4)));
typedef unsigned int uint32x4 __attribute__((ext_vector_type(4)));
typedef int       int32x4  __attribute__((ext_vector_type(4)));
typedef int       int32x8  __attribute__((ext_vector_type(8)));

#if defined(__HIP_DEVICE_COMPILE__) && __has_builtin(__builtin_amdgcn_tensor_load_to_lds) && __has_builtin(__builtin_amdgcn_s_wait_tensorcnt)
#define TDM_OK 1
#else
#define TDM_OK 0
#endif

// ---------------- helpers ----------------

__device__ __forceinline__ uint32_t pack2bf(float lo, float hi) {
#if defined(__HIP_DEVICE_COMPILE__)
    // single v_perm_b32: result = { hi[31:16], lo[31:16] }  (bf16 truncation)
    return __builtin_amdgcn_perm(__float_as_uint(hi), __float_as_uint(lo), 0x07060302u);
#else
    union { float f; uint32_t u; } a, b; a.f = lo; b.f = hi;
    return (a.u >> 16) | (b.u & 0xFFFF0000u);
#endif
}

__device__ __forceinline__ float fast_sigmoid(float x) {
    return 1.0f / (1.0f + __expf(-x));
}

// =====================================================================
// Phase 1: fp32 -> bf16 pre-conversion into d_ws
//   Abf [8192][2048] = concat(x, hx) along K
//   Wbf [4096][2048] row g*1024+h = concat(wx[g][h,:], wh[g][h,:])
// =====================================================================
struct ConvParams {
    const float* x; const float* hx;
    const float* wx[4]; const float* wh[4];
    __bf16* dst;     // Abf followed by Wbf
};

__global__ __launch_bounds__(256)
void convert_to_bf16(ConvParams C) {
    const size_t gid = (size_t)blockIdx.x * 256 + threadIdx.x;
    const size_t e   = gid * 8;                 // 8 elements per thread
    const int row = (int)(e >> 11);             // 2048 elems per row
    const int col = (int)(e & 2047);
    const float* src;
    if (row < B_DIM) {
        src = (col < IN_DIM) ? C.x  + (size_t)row * IN_DIM + col
                             : C.hx + (size_t)row * H_DIM + (col - IN_DIM);
    } else {
        const int wr = row - B_DIM;
        const int g  = wr >> 10;
        const int h  = wr & 1023;
        src = (col < IN_DIM) ? C.wx[g] + (size_t)h * IN_DIM + col
                             : C.wh[g] + (size_t)h * H_DIM + (col - IN_DIM);
    }
    const float4 f0 = ((const float4*)src)[0];
    const float4 f1 = ((const float4*)src)[1];
    u32x4 o = { pack2bf(f0.x, f0.y), pack2bf(f0.z, f0.w),
                pack2bf(f1.x, f1.y), pack2bf(f1.z, f1.w) };
    ((u32x4*)(C.dst + e))[0] = o;
}

// =====================================================================
// Phase 2: bf16 GEMM + fused LSTM epilogue, TDM-staged tiles.
//   Block tile: 256 batch rows x 32 h-cols x 4 gates. 8 waves (4M x 2H).
// =====================================================================
#define G_TILE_M 256
#define G_TILE_H 32

struct GemmParams {
    const __bf16* A;          // [8192][2048]
    const __bf16* W;          // [4096][2048]
    const float* cx;
    const float* bx[4]; const float* bh[4];
    float* hy; float* cy;
};

#if TDM_OK
// Issue one TDM tile load (2D when tile_d2==0, 3D otherwise).
__device__ __forceinline__ void tdm_tile_load(uint32_t lds_addr, uint64_t gaddr,
                                              uint32_t tile_d0, uint32_t tile_d1,
                                              uint32_t tile_d2,
                                              uint64_t stride0, uint64_t stride1,
                                              uint32_t tdim2) {
    uint32x4 g0;
    g0.x = 1u;                                   // count=1, is_restore=0, gather off
    g0.y = lds_addr;                             // LDS byte address
    g0.z = (uint32_t)gaddr;                      // global_addr[31:0]
    g0.w = (uint32_t)((gaddr >> 32) & 0x01FFFFFFu) | (2u << 30);  // [56:32] + type=2
    const uint32_t td0 = (uint32_t)K_TOT;        // tensor_dim0
    const uint32_t td1 = 0x00100000u;            // tensor_dim1 (large, no OOB)
    int32x8 g1;
    g1[0] = (int)((1u << 16)        // data_size = 2 bytes
                | (1u << 20)        // pad_enable (LDS row padding)
                | (3u << 22)        // pad_interval: every 16 DWORDs (64B row)
                | (3u << 25));      // pad_amount: 4 DWORDs (16B pad)
    g1[1] = (int)((td0 & 0xFFFFu) << 16);                         // td0 lo
    g1[2] = (int)((td0 >> 16) | ((td1 & 0xFFFFu) << 16));         // td0 hi | td1 lo
    g1[3] = (int)((td1 >> 16) | (tile_d0 << 16));                 // td1 hi | tile_dim0
    g1[4] = (int)(tile_d1 | (tile_d2 << 16));                     // tile_dim1 | tile_dim2
    g1[5] = (int)(stride0 & 0xFFFFFFFFu);                         // dim0 stride lo
    g1[6] = (int)(((stride0 >> 32) & 0xFFFFu) |
                  ((uint32_t)(stride1 & 0xFFFFu) << 16));         // dim0 hi | dim1 lo
    g1[7] = (int)((stride1 >> 16) & 0xFFFFFFFFu);                 // dim1 stride hi
    int32x4 g2 = { (int)tdim2, 0, 0, 0 };
    int32x4 g3 = { 0, 0, 0, 0 };
#if __clang_major__ >= 23
    int32x8 g4 = { 0, 0, 0, 0, 0, 0, 0, 0 };     // 6-arg toolchain form
    __builtin_amdgcn_tensor_load_to_lds(g0, g1, g2, g3, g4, 0);
#else
    __builtin_amdgcn_tensor_load_to_lds(g0, g1, g2, g3, 0);      // 5-arg ROCm form
#endif
}
#endif

__global__ __launch_bounds__(256, 2)
void lstm_gemm_bf16(GemmParams P) {
    __shared__ __align__(16) __bf16 As[2][G_TILE_M * LDS_STRIDE];      // 2 x 20480 B
    __shared__ __align__(16) __bf16 Bs[2][4 * G_TILE_H * LDS_STRIDE];  // 2 x 10240 B

    const int tid   = threadIdx.x;
    const int lane  = tid & 31;
    const int wave  = tid >> 5;
    const int waveM = wave >> 1;          // 0..3 -> 64-row strip
    const int waveH = wave & 1;           // 0..1 -> 16-col strip
    const int b0    = blockIdx.x * G_TILE_M;
    const int h0    = blockIdx.y * G_TILE_H;

    auto stage = [&](int buf, int ks) {
        const int kk = ks * TILE_K;
#if TDM_OK
        if (wave == 0) {
            const uint64_t aG = (uint64_t)(uintptr_t)P.A + (((uint64_t)b0 * K_TOT + kk) << 1);
            const uint64_t wG = (uint64_t)(uintptr_t)P.W + (((uint64_t)h0 * K_TOT + kk) << 1);
            const uint32_t aL = (uint32_t)(uintptr_t)&As[buf][0];
            const uint32_t bL = (uint32_t)(uintptr_t)&Bs[buf][0];
            // A: 2D tile 256 rows x 32 bf16
            tdm_tile_load(aL, aG, TILE_K, G_TILE_M, 0, K_TOT, 0, 0);
            // B: 3D tile 4 gates x 32 h-rows x 32 bf16 (gate stride = 1024 rows)
            tdm_tile_load(bL, wG, TILE_K, G_TILE_H, 4, K_TOT,
                          (uint64_t)H_DIM * K_TOT, 4);
        }
#else
        // fallback: cooperative bf16 copy (no conversion needed)
        {
            const u32x4* ga = (const u32x4*)(P.A + ((size_t)(b0 + tid) * K_TOT + kk));
            u32x4* la = (u32x4*)&As[buf][tid * LDS_STRIDE];
            #pragma unroll
            for (int c = 0; c < 4; ++c) la[c] = ga[c];
            const int brow = tid & 127;           // g*32 + hr
            const int pr   = tid >> 7;            // chunk pair
            const int g    = brow >> 5;
            const int hr   = brow & 31;
            const u32x4* gb = (const u32x4*)(P.W +
                ((size_t)(g * H_DIM + h0 + hr) * K_TOT + kk + pr * 16));
            u32x4* lb = (u32x4*)&Bs[buf][brow * LDS_STRIDE + pr * 16];
            lb[0] = gb[0];
            lb[1] = gb[1];
        }
#endif
    };

    v8f acc[4][4] = {};                   // [m-tile][gate]
    const int fr = lane & 15;
    const int fh = lane >> 4;

    auto compute = [&](int buf) {
        union FragU { v16bf v; u32x4 q[2]; } a[4], b[4];
        #pragma unroll
        for (int g = 0; g < 4; ++g) {
            const __bf16* p = &Bs[buf][(g * G_TILE_H + waveH * 16 + fr) * LDS_STRIDE + fh * 8];
            b[g].q[0] = *(const u32x4*)p;
            b[g].q[1] = *(const u32x4*)(p + 16);
        }
        #pragma unroll
        for (int mt = 0; mt < 4; ++mt) {
            const __bf16* p = &As[buf][(waveM * 64 + mt * 16 + fr) * LDS_STRIDE + fh * 8];
            a[mt].q[0] = *(const u32x4*)p;
            a[mt].q[1] = *(const u32x4*)(p + 16);
        }
        #pragma unroll
        for (int mt = 0; mt < 4; ++mt)
            #pragma unroll
            for (int g = 0; g < 4; ++g)
                acc[mt][g] = __builtin_amdgcn_wmma_f32_16x16x32_bf16(
                    false, a[mt].v, false, b[g].v, (short)0, acc[mt][g], false, false);
    };

    stage(0, 0);
    for (int ks = 0; ks < KSTEPS; ++ks) {
        if (ks < KSTEPS - 1) stage((ks + 1) & 1, ks + 1);
#if TDM_OK
        if (wave == 0) {
            if (ks < KSTEPS - 1) __builtin_amdgcn_s_wait_tensorcnt(2);  // stage ks landed
            else                 __builtin_amdgcn_s_wait_tensorcnt(0);
        }
#endif
        __syncthreads();
        compute(ks & 1);
        __syncthreads();
    }

    // ---- fused LSTM epilogue ----
    const int hcol = h0 + waveH * 16 + fr;
    float bsum[4];
    #pragma unroll
    for (int g = 0; g < 4; ++g)
        bsum[g] = P.bx[g][hcol] + P.bh[g][hcol];

    #pragma unroll
    for (int mt = 0; mt < 4; ++mt) {
        #pragma unroll
        for (int j = 0; j < 8; ++j) {
            const int row = b0 + waveM * 64 + mt * 16 + fh * 8 + j;
            const size_t idx = (size_t)row * H_DIM + hcol;
            const float cxv = P.cx[idx];
            const float iv = fast_sigmoid(acc[mt][0][j] + bsum[0]);
            const float fv = fast_sigmoid(acc[mt][1][j] + bsum[1]);
            const float gv = tanhf(acc[mt][2][j] + bsum[2]);
            const float ov = fast_sigmoid(acc[mt][3][j] + bsum[3]);
            const float cyv = cxv * fv + iv * gv;
            P.cy[idx] = cyv;
            P.hy[idx] = ov * tanhf(cyv);
        }
    }
}

// =====================================================================
// Fallback: single fused kernel with on-the-fly conversion (ws too small)
// =====================================================================
#define F_TILE_M 128
#define F_TILE_H 32

struct Params {
    const float* x;  const float* hx; const float* cx;
    const float* wx[4]; const float* bx[4];
    const float* wh[4]; const float* bh[4];
    float* hy; float* cy;
};

__global__ __launch_bounds__(256, 2)
void lstm_fused_fallback(Params P) {
    __shared__ __align__(16) __bf16 As[2][F_TILE_M * LDS_STRIDE];
    __shared__ __align__(16) __bf16 Bs[2][4 * F_TILE_H * LDS_STRIDE];

    const int tid   = threadIdx.x;
    const int lane  = tid & 31;
    const int wave  = tid >> 5;
    const int waveM = wave >> 1;
    const int waveH = wave & 1;
    const int b0    = blockIdx.x * F_TILE_M;
    const int h0    = blockIdx.y * F_TILE_H;

    const int srow  = tid >> 1;
    const int shalf = tid & 1;
    const int sgate = srow >> 5;
    const int shr   = srow & 31;

    uint32_t stA[8], stB[8];

    auto load_regs = [&](int ks) {
        const int phase = ks >> 5;
        const int kk    = (ks & 31) * TILE_K;
        const float* asrc = phase ? P.hx : P.x;
        const float4* ap  = (const float4*)(asrc + (size_t)(b0 + srow) * IN_DIM + kk + shalf * 16);
        const float* wsrc = (phase ? P.wh : P.wx)[sgate];
        const float4* bp  = (const float4*)(wsrc + (size_t)(h0 + shr) * IN_DIM + kk + shalf * 16);
        #pragma unroll
        for (int i = 0; i < 4; ++i) {
            float4 a = ap[i];
            stA[2*i+0] = pack2bf(a.x, a.y);
            stA[2*i+1] = pack2bf(a.z, a.w);
        }
        #pragma unroll
        for (int i = 0; i < 4; ++i) {
            float4 b = bp[i];
            stB[2*i+0] = pack2bf(b.x, b.y);
            stB[2*i+1] = pack2bf(b.z, b.w);
        }
    };

    auto store_lds = [&](int buf) {
        uint32_t* ad = (uint32_t*)&As[buf][srow * LDS_STRIDE + shalf * 16];
        u32x4 a0 = { stA[0], stA[1], stA[2], stA[3] };
        u32x4 a1 = { stA[4], stA[5], stA[6], stA[7] };
        ((u32x4*)ad)[0]       = a0;
        ((u32x4*)(ad + 4))[0] = a1;
        uint32_t* bd = (uint32_t*)&Bs[buf][(sgate * F_TILE_H + shr) * LDS_STRIDE + shalf * 16];
        u32x4 b0v = { stB[0], stB[1], stB[2], stB[3] };
        u32x4 b1v = { stB[4], stB[5], stB[6], stB[7] };
        ((u32x4*)bd)[0]       = b0v;
        ((u32x4*)(bd + 4))[0] = b1v;
    };

    v8f acc[2][4] = {};
    const int fr = lane & 15;
    const int fh = lane >> 4;

    auto compute = [&](int buf) {
        union FragU { v16bf v; u32x4 q[2]; } a[2], b[4];
        #pragma unroll
        for (int mt = 0; mt < 2; ++mt) {
            const __bf16* p = &As[buf][(waveM * 32 + mt * 16 + fr) * LDS_STRIDE + fh * 8];
            a[mt].q[0] = *(const u32x4*)p;
            a[mt].q[1] = *(const u32x4*)(p + 16);
        }
        #pragma unroll
        for (int g = 0; g < 4; ++g) {
            const __bf16* p = &Bs[buf][(g * F_TILE_H + waveH * 16 + fr) * LDS_STRIDE + fh * 8];
            b[g].q[0] = *(const u32x4*)p;
            b[g].q[1] = *(const u32x4*)(p + 16);
        }
        #pragma unroll
        for (int mt = 0; mt < 2; ++mt)
            #pragma unroll
            for (int g = 0; g < 4; ++g)
                acc[mt][g] = __builtin_amdgcn_wmma_f32_16x16x32_bf16(
                    false, a[mt].v, false, b[g].v, (short)0, acc[mt][g], false, false);
    };

    load_regs(0);
    store_lds(0);
    __syncthreads();
    for (int ks = 0; ks < KSTEPS; ++ks) {
        if (ks < KSTEPS - 1) load_regs(ks + 1);
        compute(ks & 1);
        if (ks < KSTEPS - 1) {
            store_lds((ks + 1) & 1);
            __syncthreads();
        }
    }

    const int hcol = h0 + waveH * 16 + fr;
    float bsum[4];
    #pragma unroll
    for (int g = 0; g < 4; ++g)
        bsum[g] = P.bx[g][hcol] + P.bh[g][hcol];

    #pragma unroll
    for (int mt = 0; mt < 2; ++mt) {
        #pragma unroll
        for (int j = 0; j < 8; ++j) {
            const int row = b0 + waveM * 32 + mt * 16 + fh * 8 + j;
            const size_t idx = (size_t)row * H_DIM + hcol;
            const float cxv = P.cx[idx];
            const float iv = fast_sigmoid(acc[mt][0][j] + bsum[0]);
            const float fv = fast_sigmoid(acc[mt][1][j] + bsum[1]);
            const float gv = tanhf(acc[mt][2][j] + bsum[2]);
            const float ov = fast_sigmoid(acc[mt][3][j] + bsum[3]);
            const float cyv = cxv * fv + iv * gv;
            P.cy[idx] = cyv;
            P.hy[idx] = ov * tanhf(cyv);
        }
    }
}

// =====================================================================
extern "C" void kernel_launch(void* const* d_in, const int* in_sizes, int n_in,
                              void* d_out, int out_size, void* d_ws, size_t ws_size,
                              hipStream_t stream) {
    (void)in_sizes; (void)n_in; (void)out_size;
    const float* x  = (const float*)d_in[0];
    const float* hx = (const float*)d_in[1];
    const float* cx = (const float*)d_in[2];
    const float* wx[4] = { (const float*)d_in[4], (const float*)d_in[6],
                           (const float*)d_in[8], (const float*)d_in[10] };
    const float* bx[4] = { (const float*)d_in[5], (const float*)d_in[7],
                           (const float*)d_in[9], (const float*)d_in[11] };
    const float* wh[4] = { (const float*)d_in[12], (const float*)d_in[14],
                           (const float*)d_in[16], (const float*)d_in[18] };
    const float* bh[4] = { (const float*)d_in[13], (const float*)d_in[15],
                           (const float*)d_in[17], (const float*)d_in[19] };
    float* hy = (float*)d_out;
    float* cy = (float*)d_out + (size_t)B_DIM * H_DIM;

    const size_t conv_bytes = ((size_t)B_DIM + 4 * H_DIM) * K_TOT * sizeof(__bf16); // 48 MB

    if (ws_size >= conv_bytes) {
        __bf16* Abf = (__bf16*)d_ws;
        __bf16* Wbf = Abf + (size_t)B_DIM * K_TOT;

        ConvParams C;
        C.x = x; C.hx = hx; C.dst = Abf;
        for (int g = 0; g < 4; ++g) { C.wx[g] = wx[g]; C.wh[g] = wh[g]; }
        const int conv_blocks = (int)((((size_t)B_DIM + 4 * H_DIM) * K_TOT / 8) / 256);
        convert_to_bf16<<<conv_blocks, 256, 0, stream>>>(C);

        GemmParams G;
        G.A = Abf; G.W = Wbf; G.cx = cx;
        for (int g = 0; g < 4; ++g) { G.bx[g] = bx[g]; G.bh[g] = bh[g]; }
        G.hy = hy; G.cy = cy;
        dim3 grid(B_DIM / G_TILE_M, H_DIM / G_TILE_H);   // 32 x 32
        lstm_gemm_bf16<<<grid, 256, 0, stream>>>(G);
    } else {
        Params P;
        P.x = x; P.hx = hx; P.cx = cx;
        for (int g = 0; g < 4; ++g) { P.wx[g] = wx[g]; P.bx[g] = bx[g];
                                      P.wh[g] = wh[g]; P.bh[g] = bh[g]; }
        P.hy = hy; P.cy = cy;
        dim3 grid(B_DIM / F_TILE_M, H_DIM / F_TILE_H);   // 64 x 32
        lstm_fused_fallback<<<grid, 256, 0, stream>>>(P);
    }
}
